// SparseMoETeacher_66022237274194
// MI455X (gfx1250) — matrix operand order repaired and checked by reading, not compiled, not run on hardware
//
#include <hip/hip_runtime.h>

#define N_TOK 4096
#define DIM   4096
#define NEXP  8
#define CAP   4096

#define BM 128
#define BN 64
#define KC 64
#define ASTR 72              // padded LDS row stride in bf16 elems (144B, odd multiple of 16B)
#define NCHUNK (DIM / KC)    // 64

typedef __bf16 v16bf __attribute__((ext_vector_type(16)));
typedef float  v8f   __attribute__((ext_vector_type(8)));

union FragU { uint4 q[2]; v16bf v; };

__device__ __forceinline__ unsigned short f2bf(float f) {
    unsigned int u = __float_as_uint(f);
    unsigned int r = u + 0x7FFFu + ((u >> 16) & 1u);   // round-to-nearest-even to bf16
    return (unsigned short)(r >> 16);
}
__device__ __forceinline__ float bf2f(unsigned short h) {
    return __uint_as_float(((unsigned int)h) << 16);
}

// ---------------------------------------------------------------- counters
__global__ void moe_zero_cnt(int* __restrict__ cnt) {
    if (threadIdx.x < NEXP) cnt[threadIdx.x] = 0;
}

// ---------------------------------------------------------------- router
__global__ __launch_bounds__(256)
void moe_router(const float* __restrict__ x, const float* __restrict__ Wr,
                const float* __restrict__ br, float* __restrict__ logits,
                int* __restrict__ cnt, int* __restrict__ tokl,
                float* __restrict__ wgtl) {
    const int n = blockIdx.x * 256 + threadIdx.x;
    float acc[NEXP];
#pragma unroll
    for (int e = 0; e < NEXP; ++e) acc[e] = 0.f;
    const float* xr = x + (size_t)n * DIM;
    for (int d = 0; d < DIM; d += 4) {
        float4 xv = *(const float4*)(xr + d);
#pragma unroll
        for (int e = 0; e < NEXP; ++e) {          // Wr address is wave-uniform -> SMEM path
            float4 wv = *(const float4*)(Wr + (size_t)e * DIM + d);
            acc[e] = fmaf(xv.x, wv.x, fmaf(xv.y, wv.y, fmaf(xv.z, wv.z, fmaf(xv.w, wv.w, acc[e]))));
        }
    }
    float mx = -3.0e38f;
#pragma unroll
    for (int e = 0; e < NEXP; ++e) {
        acc[e] += br[e];
        logits[(size_t)n * NEXP + e] = acc[e];
        mx = fmaxf(mx, acc[e]);
    }
    float p[NEXP], s = 0.f;
#pragma unroll
    for (int e = 0; e < NEXP; ++e) { p[e] = expf(acc[e] - mx); s += p[e]; }
    const float inv = 1.0f / s;
    // top-2 (earliest index wins ties, matching jax top_k)
    int i0 = 0; float v0 = p[0];
#pragma unroll
    for (int e = 1; e < NEXP; ++e) if (p[e] > v0) { v0 = p[e]; i0 = e; }
    int i1 = -1; float v1 = -1.f;
#pragma unroll
    for (int e = 0; e < NEXP; ++e) if (e != i0 && p[e] > v1) { v1 = p[e]; i1 = e; }

    int s0 = atomicAdd(&cnt[i0], 1);
    tokl[i0 * CAP + s0] = n;                 // k = 0
    wgtl[i0 * CAP + s0] = v0 * inv;
    int s1 = atomicAdd(&cnt[i1], 1);
    tokl[i1 * CAP + s1] = n | (1 << 16);     // k = 1
    wgtl[i1 * CAP + s1] = v1 * inv;
}

// ---------------------------------------------------------------- grouped expert GEMM (bf16x3 split)
__global__ __launch_bounds__(256)
void moe_expert_gemm(const float* __restrict__ x, const float* __restrict__ We,
                     const float* __restrict__ be, const int* __restrict__ cnt,
                     const int* __restrict__ tokl, const float* __restrict__ wgtl,
                     float* __restrict__ out0, float* __restrict__ out1) {
    const int e      = blockIdx.z;
    const int m_base = blockIdx.y * BM;
    const int n_base = blockIdx.x * BN;
    const int count  = cnt[e];
    if (m_base >= count) return;

    __shared__ __align__(16) unsigned short Axh[BM * ASTR];
    __shared__ __align__(16) unsigned short Axl[BM * ASTR];
    __shared__ __align__(16) unsigned short Bwh[BN * ASTR];
    __shared__ __align__(16) unsigned short Bwl[BN * ASTR];

    const int tid = threadIdx.x;

    // global-load mapping: X tile 128x64 f32 (8 float4/thr), W tile 64x64 f32 (4 float4/thr)
    const int xrow = tid >> 1;
    const int xcol = (tid & 1) * 32;
    const int wrow = tid >> 2;
    const int wcol = (tid & 3) * 16;

    const int row_g = m_base + xrow;
    const bool rok  = (row_g < count);
    const int slotv = tokl[e * CAP + (rok ? row_g : m_base)];
    const int tok   = rok ? (slotv & 0xFFFF) : 0;

    const float* gx = x  + (size_t)tok * DIM + xcol;
    const float* gw = We + (size_t)e * DIM * DIM + (size_t)(n_base + wrow) * DIM + wcol;

    float4 rx[8], rw[4];
#pragma unroll
    for (int i = 0; i < 8; ++i) rx[i] = *(const float4*)(gx + 4 * i);
#pragma unroll
    for (int i = 0; i < 4; ++i) rw[i] = *(const float4*)(gw + 4 * i);

    // wave tiling: 8 waves, wave owns M-tiles {wm, wm+4}, N-tiles {2*wn, 2*wn+1}
    const int wid  = tid >> 5;
    const int lane = tid & 31;
    const int half = lane >> 4;
    const int r16  = lane & 15;
    const int wm   = wid & 3;
    const int wn   = wid >> 2;

    v8f acc[2][2];
#pragma unroll
    for (int m = 0; m < 2; ++m)
#pragma unroll
        for (int j = 0; j < 2; ++j)
            acc[m][j] = (v8f){0.f, 0.f, 0.f, 0.f, 0.f, 0.f, 0.f, 0.f};

    for (int kc = 0; kc < NCHUNK; ++kc) {
        __syncthreads();   // previous chunk's LDS fully consumed
        // fp32 -> (bf16 hi, bf16 lo) conversion + LDS staging
#pragma unroll
        for (int i = 0; i < 8; ++i) {
            float4 v = rx[i];
            unsigned short h0 = f2bf(v.x), h1 = f2bf(v.y), h2 = f2bf(v.z), h3 = f2bf(v.w);
            unsigned short l0 = f2bf(v.x - bf2f(h0)), l1 = f2bf(v.y - bf2f(h1));
            unsigned short l2 = f2bf(v.z - bf2f(h2)), l3 = f2bf(v.w - bf2f(h3));
            const int off = xrow * ASTR + xcol + 4 * i;
            uint2 ph; ph.x = (unsigned)h0 | ((unsigned)h1 << 16); ph.y = (unsigned)h2 | ((unsigned)h3 << 16);
            uint2 pl; pl.x = (unsigned)l0 | ((unsigned)l1 << 16); pl.y = (unsigned)l2 | ((unsigned)l3 << 16);
            *(uint2*)(Axh + off) = ph;
            *(uint2*)(Axl + off) = pl;
        }
#pragma unroll
        for (int i = 0; i < 4; ++i) {
            float4 v = rw[i];
            unsigned short h0 = f2bf(v.x), h1 = f2bf(v.y), h2 = f2bf(v.z), h3 = f2bf(v.w);
            unsigned short l0 = f2bf(v.x - bf2f(h0)), l1 = f2bf(v.y - bf2f(h1));
            unsigned short l2 = f2bf(v.z - bf2f(h2)), l3 = f2bf(v.w - bf2f(h3));
            const int off = wrow * ASTR + wcol + 4 * i;
            uint2 ph; ph.x = (unsigned)h0 | ((unsigned)h1 << 16); ph.y = (unsigned)h2 | ((unsigned)h3 << 16);
            uint2 pl; pl.x = (unsigned)l0 | ((unsigned)l1 << 16); pl.y = (unsigned)l2 | ((unsigned)l3 << 16);
            *(uint2*)(Bwh + off) = ph;
            *(uint2*)(Bwl + off) = pl;
        }
        __syncthreads();   // LDS visible

        if (kc + 1 < NCHUNK) {   // software prefetch of next K-chunk (overlaps WMMA below)
            const float* nx = gx + (size_t)(kc + 1) * KC;
            const float* nw = gw + (size_t)(kc + 1) * KC;
#pragma unroll
            for (int i = 0; i < 8; ++i) rx[i] = *(const float4*)(nx + 4 * i);
#pragma unroll
            for (int i = 0; i < 4; ++i) rw[i] = *(const float4*)(nw + 4 * i);
        }

#pragma unroll
        for (int ks = 0; ks < 2; ++ks) {
            // A fragments: lane half h reads K {h*8..h*8+7} and {16+h*8..}, per ISA A-layout
            FragU axh[2], axl[2];
#pragma unroll
            for (int m = 0; m < 2; ++m) {
                const int rowA = (wm + 4 * m) * 16 + r16;
                const int o = rowA * ASTR + ks * 32 + half * 8;
                axh[m].q[0] = *(const uint4*)(Axh + o);
                axh[m].q[1] = *(const uint4*)(Axh + o + 16);
                axl[m].q[0] = *(const uint4*)(Axl + o);
                axl[m].q[1] = *(const uint4*)(Axl + o + 16);
            }
#pragma unroll
            for (int j = 0; j < 2; ++j) {
                // B fragments: lane half h reads K {h*8..h*8+7} and {16+h*8..}, matching A
                const int rowB = (wn * 2 + j) * 16 + r16;
                const int o = rowB * ASTR + ks * 32 + half * 8;
                FragU bh, bl;
                bh.q[0] = *(const uint4*)(Bwh + o);
                bh.q[1] = *(const uint4*)(Bwh + o + 16);
                bl.q[0] = *(const uint4*)(Bwl + o);
                bl.q[1] = *(const uint4*)(Bwl + o + 16);
#pragma unroll
                for (int m = 0; m < 2; ++m) {
                    acc[m][j] = __builtin_amdgcn_wmma_f32_16x16x32_bf16(
                        false, axh[m].v, false, bh.v, (short)0, acc[m][j], false, false);
                    acc[m][j] = __builtin_amdgcn_wmma_f32_16x16x32_bf16(
                        false, axh[m].v, false, bl.v, (short)0, acc[m][j], false, false);
                    acc[m][j] = __builtin_amdgcn_wmma_f32_16x16x32_bf16(
                        false, axl[m].v, false, bh.v, (short)0, acc[m][j], false, false);
                }
            }
        }
    }

    // epilogue: scale by routing weight, add bias, route k=0 -> out0 (store), k=1 -> out1 (store)
#pragma unroll
    for (int m = 0; m < 2; ++m) {
        const int trow0 = (wm + 4 * m) * 16 + 8 * half;
#pragma unroll
        for (int j = 0; j < 2; ++j) {
            const int col = n_base + (wn * 2 + j) * 16 + r16;
            const float bias = be[(size_t)e * DIM + col];
#pragma unroll
            for (int rr = 0; rr < 8; ++rr) {
                const int rg = m_base + trow0 + rr;
                if (rg < count) {
                    const int sv = tokl[e * CAP + rg];
                    const int tk = sv & 0xFFFF;
                    const float w = wgtl[e * CAP + rg];
                    const float val = w * (acc[m][j][rr] + bias);
                    float* dst = (sv & (1 << 16)) ? out1 : out0;
                    dst[(size_t)tk * DIM + col] = val;
                }
            }
        }
    }
}

// ---------------------------------------------------------------- combine k=0 + k=1 planes
__global__ void moe_combine(float* __restrict__ out, const float* __restrict__ buf, int n4) {
    const int i = blockIdx.x * 256 + threadIdx.x;
    if (i < n4) {
        float4 a = ((const float4*)out)[i];
        float4 b = ((const float4*)buf)[i];
        a.x += b.x; a.y += b.y; a.z += b.z; a.w += b.w;
        ((float4*)out)[i] = a;
    }
}

extern "C" void kernel_launch(void* const* d_in, const int* in_sizes, int n_in,
                              void* d_out, int out_size, void* d_ws, size_t ws_size,
                              hipStream_t stream) {
    (void)in_sizes; (void)n_in; (void)out_size; (void)ws_size;
    const float* x  = (const float*)d_in[0];
    const float* Wr = (const float*)d_in[1];
    const float* br = (const float*)d_in[2];
    const float* We = (const float*)d_in[3];
    const float* be = (const float*)d_in[4];

    float* out    = (float*)d_out;                       // [N*D] output
    float* logits = out + (size_t)N_TOK * DIM;           // [N*E] logits

    char*  ws   = (char*)d_ws;
    int*   cnt  = (int*)ws;                              // 8 ints (256B reserved)
    int*   tokl = (int*)(ws + 256);                      // E*CAP ints
    float* wgtl = (float*)(ws + 256 + NEXP * CAP * 4);   // E*CAP floats
    float* buf  = (float*)(ws + 256 + 2 * NEXP * CAP * 4); // N*D floats (k=1 plane)

    moe_zero_cnt<<<1, 32, 0, stream>>>(cnt);
    moe_router<<<N_TOK / 256, 256, 0, stream>>>(x, Wr, br, logits, cnt, tokl, wgtl);
    dim3 grid(DIM / BN, CAP / BM, NEXP);
    moe_expert_gemm<<<grid, 256, 0, stream>>>(x, We, be, cnt, tokl, wgtl, out, buf);
    moe_combine<<<(N_TOK * DIM / 4 + 255) / 256, 256, 0, stream>>>(out, buf, N_TOK * DIM / 4);
}
